// GNNRoutePlanner_39926015983752
// MI455X (gfx1250) — matrix operand order repaired and checked by reading, not compiled, not run on hardware
//
#include <hip/hip_runtime.h>

typedef __attribute__((ext_vector_type(16))) _Float16 v16h;
typedef __attribute__((ext_vector_type(8)))  float    v8f;

#define CHID 64

__device__ __forceinline__ float leaky01(float v) { return v >= 0.f ? v : 0.01f * v; }

// ---------- degree / normalization ----------
__global__ void k_init_deg(unsigned* deg, int n) {
  int i = blockIdx.x * blockDim.x + threadIdx.x;
  if (i < n) deg[i] = 1u;                       // self-loop
}

__global__ void k_count_deg(const long long* __restrict__ ei, unsigned* deg, int E) {
  int i = blockIdx.x * blockDim.x + threadIdx.x;
  if (i < E) atomicAdd(&deg[(int)ei[(size_t)E + i]], 1u);   // dst occurrences
}

__global__ void k_dis(float* dis, int n) {      // in-place u32 -> rsqrt(f32)
  int i = blockIdx.x * blockDim.x + threadIdx.x;
  if (i < n) { unsigned u = ((const unsigned*)dis)[i]; dis[i] = rsqrtf((float)u); }
}

// ---------- pre-swizzle W[K,64] fp32 -> CDNA5 WMMA B-fragment f16 layout ----------
// fragment (kk,nt): v16h per lane; lanes 0-15: col=lane, K=kk*32+{0..7,16..23} ; lanes 16-31: col=lane-16, K=kk*32+{8..15,24..31}
__global__ void k_swizzle_w(const float* __restrict__ W, _Float16* __restrict__ out, int Kdim) {
  int t = blockIdx.x * blockDim.x + threadIdx.x;
  int total = (Kdim >> 5) * 4 * 32 * 16;
  if (t >= total) return;
  int j = t & 15, lane = (t >> 4) & 31, nt = (t >> 9) & 3, kk = t >> 11;
  int k = kk * 32 + ((lane >> 4) << 3) + (j & 7) + ((j >> 3) << 4);
  int n = nt * 16 + (lane & 15);
  out[t] = (_Float16)W[(size_t)k * CHID + n];
}

// ---------- GEMM: Out[nrows,64] = A[nrows,Kdim] @ W (wave = 16x64 tile) ----------
__global__ void __launch_bounds__(256)
k_gemm(const float* __restrict__ A, const _Float16* __restrict__ Wf,
       float* __restrict__ Out, int Kdim, int nrows, int apply_leaky) {
  int wid  = (int)((blockIdx.x * (unsigned)blockDim.x + threadIdx.x) >> 5);
  int lane = threadIdx.x & 31;
  if (wid >= (nrows >> 4)) return;              // wave-uniform: EXEC stays all-ones
  int rowbase = wid << 4;
  int hi = lane >> 4;                           // 0 = lanes 0-15, 1 = lanes 16-31
  const float* arow = A + (size_t)(rowbase + (lane & 15)) * Kdim;
  const v16h* B = (const v16h*)Wf;
  v8f acc0 = {}, acc1 = {}, acc2 = {}, acc3 = {};
  int ksteps = Kdim >> 5;
  for (int kk = 0; kk < ksteps; ++kk) {
    // A fragment: lanes<16 hold K = k0+{0..7,16..23}; lanes>=16 shifted by 8
    int k0 = (kk << 5) + (hi << 3);
    v16h a;
#pragma unroll
    for (int j = 0; j < 8; ++j) {
      float v0 = arow[k0 + j];
      float v1 = arow[k0 + 16 + j];
      if (apply_leaky) { v0 = leaky01(v0); v1 = leaky01(v1); }
      a[j]     = (_Float16)v0;
      a[j + 8] = (_Float16)v1;
    }
    v16h b0 = B[(kk * 4 + 0) * 32 + lane];
    v16h b1 = B[(kk * 4 + 1) * 32 + lane];
    v16h b2 = B[(kk * 4 + 2) * 32 + lane];
    v16h b3 = B[(kk * 4 + 3) * 32 + lane];
    acc0 = __builtin_amdgcn_wmma_f32_16x16x32_f16(false, a, false, b0, (short)0, acc0, false, false);
    acc1 = __builtin_amdgcn_wmma_f32_16x16x32_f16(false, a, false, b1, (short)0, acc1, false, false);
    acc2 = __builtin_amdgcn_wmma_f32_16x16x32_f16(false, a, false, b2, (short)0, acc2, false, false);
    acc3 = __builtin_amdgcn_wmma_f32_16x16x32_f16(false, a, false, b3, (short)0, acc3, false, false);
  }
  // C/D layout: VGPR r, lanes0-15 -> M=r, lanes16-31 -> M=8+r, col = lane&15
#pragma unroll
  for (int r = 0; r < 8; ++r) {
    int m = rowbase + r + (hi << 3);
    float* o = Out + (size_t)m * CHID + (lane & 15);
    o[0]  = acc0[r];
    o[16] = acc1[r];
    o[32] = acc2[r];
    o[48] = acc3[r];
  }
}

// ---------- agg init: bias + self-loop term (non-atomic) ----------
__global__ void k_agg_init(const float* __restrict__ h, const float* __restrict__ dis,
                           const float* __restrict__ bias, float* __restrict__ out, int n) {
  int i = blockIdx.x * blockDim.x + threadIdx.x;
  if (i >= n * CHID) return;
  int node = i >> 6, c = i & 63;
  float d = dis[node];
  out[i] = bias[c] + h[i] * d * d;
}

// ---------- edge scatter: 16 threads per edge, 4 channels each ----------
__global__ void k_scatter(const long long* __restrict__ ei, const float* __restrict__ dis,
                          const float* __restrict__ h, float* __restrict__ out, int E) {
  long long i = (long long)blockIdx.x * blockDim.x + threadIdx.x;
  if (i >= (long long)E * 16) return;
  int e = (int)(i >> 4), q = (int)(i & 15);
  int s = (int)ei[e], d = (int)ei[(size_t)E + e];
  float nrm = dis[s] * dis[d];
  float4 v = ((const float4*)(h + (size_t)s * CHID))[q];
  float* o = out + (size_t)d * CHID + q * 4;
  unsafeAtomicAdd(o + 0, v.x * nrm);
  unsafeAtomicAdd(o + 1, v.y * nrm);
  unsafeAtomicAdd(o + 2, v.z * nrm);
  unsafeAtomicAdd(o + 3, v.w * nrm);
}

// ---------- layer 3: s3[n] = leaky(h[n,:]) . W3  (wave per node) ----------
__global__ void k_dot3(const float* __restrict__ h, const float* __restrict__ W3,
                       float* __restrict__ s3, int n) {
  int wid  = (int)((blockIdx.x * (unsigned)blockDim.x + threadIdx.x) >> 5);
  int lane = threadIdx.x & 31;
  if (wid >= n) return;
  const float* row = h + (size_t)wid * CHID;
  float acc = leaky01(row[lane]) * W3[lane] + leaky01(row[32 + lane]) * W3[32 + lane];
#pragma unroll
  for (int off = 16; off > 0; off >>= 1) acc += __shfl_xor(acc, off, 32);
  if (lane == 0) s3[wid] = acc;
}

__global__ void k_out_init(const float* __restrict__ s3, const float* __restrict__ dis,
                           const float* __restrict__ b3, float* __restrict__ out, int n) {
  int i = blockIdx.x * blockDim.x + threadIdx.x;
  if (i < n) { float d = dis[i]; out[i] = b3[0] + s3[i] * d * d; }
}

__global__ void k_scatter3(const long long* __restrict__ ei, const float* __restrict__ dis,
                           const float* __restrict__ s3, float* __restrict__ out, int E) {
  int i = blockIdx.x * blockDim.x + threadIdx.x;
  if (i >= E) return;
  int s = (int)ei[i], d = (int)ei[(size_t)E + i];
  unsafeAtomicAdd(&out[d], s3[s] * dis[s] * dis[d]);
}

extern "C" void kernel_launch(void* const* d_in, const int* in_sizes, int n_in,
                              void* d_out, int out_size, void* d_ws, size_t ws_size,
                              hipStream_t stream) {
  const float*     x  = (const float*)d_in[0];
  const long long* ei = (const long long*)d_in[1];
  const float*     W1 = (const float*)d_in[2];
  const float*     b1 = (const float*)d_in[3];
  const float*     W2 = (const float*)d_in[4];
  const float*     b2 = (const float*)d_in[5];
  const float*     W3 = (const float*)d_in[6];
  const float*     b3 = (const float*)d_in[7];
  float* out = (float*)d_out;

  const int CIN = 128;
  int N = in_sizes[0] / CIN;   // 50000 (divisible by 16)
  int E = in_sizes[1] / 2;     // 800000

  auto al = [](size_t v) { return (v + 255) & ~(size_t)255; };
  char* w = (char*)d_ws;
  size_t off = 0;
  float*    dis  = (float*)(w + off);    off = al(off + (size_t)N * 4);
  float*    bufA = (float*)(w + off);    off = al(off + (size_t)N * CHID * 4);
  float*    bufB = (float*)(w + off);    off = al(off + (size_t)N * CHID * 4);
  float*    s3   = (float*)(w + off);    off = al(off + (size_t)N * 4);
  _Float16* w1f  = (_Float16*)(w + off); off = al(off + (size_t)128 * 64 * 2);
  _Float16* w2f  = (_Float16*)(w + off); off = al(off + (size_t)64 * 64 * 2);

  const int B = 256;
  // normalization
  k_init_deg <<<(N + B - 1) / B, B, 0, stream>>>((unsigned*)dis, N);
  k_count_deg<<<(E + B - 1) / B, B, 0, stream>>>(ei, (unsigned*)dis, E);
  k_dis      <<<(N + B - 1) / B, B, 0, stream>>>(dis, N);
  // weight fragments
  k_swizzle_w<<<(8192 + B - 1) / B, B, 0, stream>>>(W1, w1f, 128);
  k_swizzle_w<<<(4096 + B - 1) / B, B, 0, stream>>>(W2, w2f, 64);

  int gemm_grid = ((N / 16) * 32 + B - 1) / B;
  int elem_grid = (N * CHID + B - 1) / B;
  long long sc  = (long long)E * 16;
  int scat_grid = (int)((sc + B - 1) / B);

  // layer 1
  k_gemm     <<<gemm_grid, B, 0, stream>>>(x, w1f, bufA, 128, N, 0);
  k_agg_init <<<elem_grid, B, 0, stream>>>(bufA, dis, b1, bufB, N);
  k_scatter  <<<scat_grid, B, 0, stream>>>(ei, dis, bufA, bufB, E);
  // layer 2 (leaky fused into GEMM A-load)
  k_gemm     <<<gemm_grid, B, 0, stream>>>(bufB, w2f, bufA, 64, N, 1);
  k_agg_init <<<elem_grid, B, 0, stream>>>(bufA, dis, b2, bufB, N);
  k_scatter  <<<scat_grid, B, 0, stream>>>(ei, dis, bufA, bufB, E);
  // layer 3 (leaky fused into dot)
  k_dot3     <<<(N * 32 + B - 1) / B, B, 0, stream>>>(bufB, W3, s3, N);
  k_out_init <<<(N + B - 1) / B, B, 0, stream>>>(s3, dis, b3, out, N);
  k_scatter3 <<<(E + B - 1) / B, B, 0, stream>>>(ei, dis, s3, out, E);
}
